// Model_26310969655986
// MI455X (gfx1250) — matrix operand order repaired and checked by reading, not compiled, not run on hardware
//
#include <hip/hip_runtime.h>
#include <stdint.h>

#define D_MAIN 256
#define D_HID  512
#define CTX    96

typedef __attribute__((ext_vector_type(16))) __bf16         v16bf;
typedef __attribute__((ext_vector_type(8)))  float          v8f;
typedef __attribute__((ext_vector_type(8)))  unsigned short us8;
typedef __attribute__((ext_vector_type(16))) unsigned short us16;
typedef __attribute__((__vector_size__(16))) int            v4i;

#if defined(__has_builtin)
#if __has_builtin(__builtin_amdgcn_global_load_async_to_lds_b128)
#define USE_ASYNC_LDS 1
#endif
#endif

#if USE_ASYNC_LDS
typedef __attribute__((address_space(1))) v4i gv4i;   // global 16B vector
typedef __attribute__((address_space(3))) v4i lv4i;   // LDS 16B vector
#endif

__device__ __forceinline__ unsigned short f2bf(float f) {
  union { float f; unsigned u; } v; v.f = f;
  unsigned r = v.u + 0x7fffu + ((v.u >> 16) & 1u);
  return (unsigned short)(r >> 16);
}
__device__ __forceinline__ float bf2f(unsigned short u) {
  union { unsigned u; float f; } v; v.u = ((unsigned)u) << 16;
  return v.f;
}

// 16B global -> LDS copy, async (ASYNCcnt) when the gfx1250 builtin exists.
__device__ __forceinline__ void cp16_g2l(const unsigned short* g, unsigned short* l) {
#if USE_ASYNC_LDS
  __builtin_amdgcn_global_load_async_to_lds_b128(
      (gv4i*)(void*)g, (lv4i*)(void*)l, 0, 0);
#else
  *(us8*)l = *(const us8*)g;
#endif
}
__device__ __forceinline__ void async_fence() {
#if USE_ASYNC_LDS
#if __has_builtin(__builtin_amdgcn_s_wait_asynccnt)
  __builtin_amdgcn_s_wait_asynccnt(0);
#else
  asm volatile("s_wait_asynccnt 0" ::: "memory");
#endif
#endif
}

// A-fragment, 16x32 bf16 (ISA 7.12.2): lanes 0-15 row M=lane, K={kb..kb+7, kb+16..kb+23};
// lanes 16-31 row M=lane-16, K={kb+8..kb+15, kb+24..kb+31}.
__device__ __forceinline__ v16bf fragA(const unsigned short* rowbase, int kb, int lane) {
  const unsigned short* p = rowbase + kb + ((lane & 16) ? 8 : 0);
  us8 lo = *(const us8*)p;
  us8 hi = *(const us8*)(p + 16);
  us16 c;
#pragma unroll
  for (int i = 0; i < 8; ++i) { c[i] = lo[i]; c[i + 8] = hi[i]; }
  return __builtin_bit_cast(v16bf, c);
}
// B-fragment, 32x16 bf16: lanes 0-15 col N=lane, K={kb..kb+7, kb+16..kb+23}; lanes 16-31 K={kb+8..kb+15, kb+24..kb+31}.
__device__ __forceinline__ v16bf fragB(const unsigned short* rowbase, int kb, int lane) {
  const unsigned short* p = rowbase + kb + ((lane & 16) ? 8 : 0);
  us8 lo = *(const us8*)p;
  us8 hi = *(const us8*)(p + 16);
  us16 c;
#pragma unroll
  for (int i = 0; i < 8; ++i) { c[i] = lo[i]; c[i + 8] = hi[i]; }
  return __builtin_bit_cast(v16bf, c);
}
#define WMMA_BF16(af, bf, acc) \
  __builtin_amdgcn_wmma_f32_16x16x32_bf16(false, (af), false, (bf), (short)0, (acc), false, false)

// ---------------------------------------------------------------- small prep kernels
__global__ void pack_feat_kernel(const float* __restrict__ xnum, const float* __restrict__ xbin,
                                 unsigned short* __restrict__ dst, int rows) {
  int i = blockIdx.x * 256 + threadIdx.x;
  if (i >= rows * 128) return;
  int r = i >> 7, j = i & 127;
  float v = (j < 96) ? xnum[r * 96 + j] : ((j < 112) ? xbin[r * 16 + (j - 96)] : 0.0f);
  dst[i] = f2bf(v);
}

__global__ void transpose_bf16_kernel(const float* __restrict__ src, unsigned short* __restrict__ dst,
                                      int K, int N, int Kpad) {
  int i = blockIdx.x * 256 + threadIdx.x;
  if (i >= N * Kpad) return;
  int n = i / Kpad, k = i - n * Kpad;
  dst[i] = (k < K) ? f2bf(src[(size_t)k * N + n]) : (unsigned short)0;
}

__global__ void rowsumsq_kernel(const unsigned short* __restrict__ a, float* __restrict__ out) {
  __shared__ float red[256];
  int r = blockIdx.x, d = threadIdx.x;
  float v = bf2f(a[(size_t)r * D_MAIN + d]);
  red[d] = v * v;
  __syncthreads();
  for (int s = 128; s > 0; s >>= 1) { if (d < s) red[d] += red[d + s]; __syncthreads(); }
  if (d == 0) out[r] = red[0];
}

// ---------------------------------------------------------------- bf16 WMMA GEMM
// Block = 8 waves. A m-tile(s) staged in LDS (async copy, padded stride for bank-free
// fragment reads, shared by all waves); each wave register-blocks a 16x64 output strip
// so one A-fragment feeds 4 WMMAs.  N must be 256 or 512; Kp multiple of 32.
// mode==1: weighted-V store: outF[row,col] = wrow[row]*(acc + yv[row]*Yw[col] + Yb[col])
__global__ __launch_bounds__(256)
void gemm_bf16_kernel(const unsigned short* __restrict__ A, const unsigned short* __restrict__ Bt,
                      const float* __restrict__ bias, const float* __restrict__ radd,
                      float* __restrict__ outF, unsigned short* __restrict__ outB,
                      int M, int N, int Kp, int relu, int mode,
                      const float* __restrict__ wrow, const float* __restrict__ yv,
                      const float* __restrict__ Yw, const float* __restrict__ Yb) {
  extern __shared__ __align__(16) unsigned short ldsA[];   // MT * 16 * (Kp+8)
  int lane = threadIdx.x & 31;
  int wave = threadIdx.x >> 5;
  int WPT = N >> 6;       // waves per m-tile (4 for N=256, 8 for N=512)
  int MT  = 8 / WPT;      // m-tiles per block (2 or 1)
  int KpL = Kp + 8;       // padded LDS stride: bank stride 4 -> conflict-free frag reads
  int mt0 = blockIdx.x * MT;

  // stage A rows [mt0*16, mt0*16 + MT*16) into LDS
  int cpr = Kp >> 3;                 // 16B chunks per row
  int totalChunks = MT * 16 * cpr;   // multiple of 256
  for (int c = threadIdx.x; c < totalChunks; c += 256) {
    int row = c / cpr;
    int off = (c - row * cpr) << 3;
    cp16_g2l(A + (size_t)(mt0 * 16 + row) * Kp + off, ldsA + row * KpL + off);
  }
  async_fence();
  __syncthreads();

  int wm = wave / WPT;               // which m-tile
  int wn = wave - wm * WPT;          // 64-col strip within N
  const unsigned short* arow  = ldsA + (wm * 16 + (lane & 15)) * KpL;
  const unsigned short* brow0 = Bt + (size_t)(wn * 64 +  0 + (lane & 15)) * Kp;
  const unsigned short* brow1 = Bt + (size_t)(wn * 64 + 16 + (lane & 15)) * Kp;
  const unsigned short* brow2 = Bt + (size_t)(wn * 64 + 32 + (lane & 15)) * Kp;
  const unsigned short* brow3 = Bt + (size_t)(wn * 64 + 48 + (lane & 15)) * Kp;

  v8f a0 = {}, a1 = {}, a2 = {}, a3 = {};
  for (int kb = 0; kb < Kp; kb += 32) {
    v16bf af = fragA(arow, kb, lane);
    v16bf b0 = fragB(brow0, kb, lane);
    v16bf b1 = fragB(brow1, kb, lane);
    v16bf b2 = fragB(brow2, kb, lane);
    v16bf b3 = fragB(brow3, kb, lane);
    a0 = WMMA_BF16(af, b0, a0);
    a1 = WMMA_BF16(af, b1, a1);
    a2 = WMMA_BF16(af, b2, a2);
    a3 = WMMA_BF16(af, b3, a3);
  }

  v8f accs[4] = {a0, a1, a2, a3};
  int mbase = (mt0 + wm) * 16 + ((lane & 16) ? 8 : 0);
#pragma unroll
  for (int s = 0; s < 4; ++s) {
    int col = wn * 64 + s * 16 + (lane & 15);
    if (mode == 0) {
      float bc = bias ? bias[col] : 0.0f;
#pragma unroll
      for (int v = 0; v < 8; ++v) {
        int m = mbase + v;
        float val = accs[s][v] + bc;
        if (radd) val += radd[(size_t)m * N + col];
        if (relu) val = val > 0.0f ? val : 0.0f;
        if (outF) outF[(size_t)m * N + col] = val;
        if (outB) outB[(size_t)m * N + col] = f2bf(val);
      }
    } else {
      float ywc = Yw[col], ybc = Yb[col];
#pragma unroll
      for (int v = 0; v < 8; ++v) {
        int m = mbase + v;
        float val = accs[s][v] + yv[m] * ywc + ybc;   // encode_y + T(diff)
        outF[(size_t)m * N + col] = wrow[m] * val;
      }
    }
  }
}

// ---------------------------------------------------------------- fused dist-GEMM + top-96
// One block = 16 batch rows. Query keys staged once in LDS (re-used for all 391 chunks);
// each wave register-blocks 16x32 scores (A-fragment shared by 2 WMMAs). Per-row streaming
// top-96 with strict (val,idx) total order => deterministic, matches top_k tie behavior.
__global__ __launch_bounds__(256)
void knn_topk_kernel(const unsigned short* __restrict__ kb16, const unsigned short* __restrict__ kib,
                     const float* __restrict__ ki2, int Ncand, int Npad,
                     int* __restrict__ topIdx, float* __restrict__ wout) {
  __shared__ __align__(16) unsigned short ldsK[16 * 264];  // padded stride
  __shared__ float sc[16][256];
  __shared__ float tV[16 * CTX];
  __shared__ int   tI[16 * CTX];
  __shared__ int   cnt[16];
  int lane = threadIdx.x & 31;
  int wave = threadIdx.x >> 5;
  int rb0 = blockIdx.x * 16;

  for (int c = threadIdx.x; c < 16 * 32; c += 256) {   // 16 rows x 32 chunks of 16B
    int row = c >> 5, off = (c & 31) << 3;
    cp16_g2l(kb16 + (size_t)(rb0 + row) * D_MAIN + off, ldsK + row * 264 + off);
  }
  if (threadIdx.x < 16) cnt[threadIdx.x] = 0;
  async_fence();
  __syncthreads();

  const unsigned short* arow = ldsK + (lane & 15) * 264;

  for (int n0 = 0; n0 < Npad; n0 += 256) {
    const unsigned short* brow0 = kib + (size_t)(n0 + wave * 32 + (lane & 15)) * D_MAIN;
    const unsigned short* brow1 = brow0 + (size_t)16 * D_MAIN;
    v8f a0 = {}, a1 = {};
    for (int kb = 0; kb < D_MAIN; kb += 32) {
      v16bf af = fragA(arow, kb, lane);
      v16bf b0 = fragB(brow0, kb, lane);
      v16bf b1 = fragB(brow1, kb, lane);
      a0 = WMMA_BF16(af, b0, a0);
      a1 = WMMA_BF16(af, b1, a1);
    }
    v8f accs[2] = {a0, a1};
    int mb = (lane & 16) ? 8 : 0;
#pragma unroll
    for (int t = 0; t < 2; ++t) {
      int cl = wave * 32 + t * 16 + (lane & 15);
      int ncol = n0 + cl;
      float kn2 = ki2[ncol];
#pragma unroll
      for (int v = 0; v < 8; ++v) {
        float s = kn2 - 2.0f * accs[t][v];
        if (ncol >= Ncand) s = 3.0e38f;   // mask padding regardless of garbage data
        sc[mb + v][cl] = s;
      }
    }
    __syncthreads();
    if (threadIdx.x < 16) {               // owner thread per row: threshold-pruned insert
      int r = threadIdx.x;
      int c = cnt[r];
      float* tv = &tV[r * CTX];
      int*   ti = &tI[r * CTX];
      for (int j = 0; j < 256; ++j) {
        float val = sc[r][j];
        int idx = n0 + j;
        if (c >= CTX) {
          float wv = tv[CTX - 1];
          if (!(val < wv || (val == wv && idx < ti[CTX - 1]))) continue;
          c = CTX - 1;
        }
        int p = c;
        while (p > 0 && (tv[p - 1] > val || (tv[p - 1] == val && ti[p - 1] > idx))) {
          tv[p] = tv[p - 1]; ti[p] = ti[p - 1]; --p;
        }
        tv[p] = val; ti[p] = idx; ++c;
      }
      cnt[r] = c;
    }
    __syncthreads();
  }

  if (threadIdx.x < 16) {                 // softmax(dist) == softmax(score): k^2 row-constant
    int r = threadIdx.x;
    float* tv = &tV[r * CTX];
    int*   ti = &tI[r * CTX];
    float mx = -3.0e38f;
    for (int j = 0; j < CTX; ++j) mx = fmaxf(mx, tv[j]);
    float s = 0.0f;
    for (int j = 0; j < CTX; ++j) { float e = __expf(tv[j] - mx); tv[j] = e; s += e; }
    float inv = 1.0f / s;
    for (int j = 0; j < CTX; ++j) {
      topIdx[(size_t)(rb0 + r) * CTX + j] = ti[j];
      wout[(size_t)(rb0 + r) * CTX + j]   = tv[j] * inv;
    }
  }
}

// ---------------------------------------------------------------- epilogue kernels
__global__ void build_diff_kernel(const float* __restrict__ kf, const unsigned short* __restrict__ kib,
                                  const int* __restrict__ topIdx, const float* __restrict__ cand_y,
                                  unsigned short* __restrict__ diff, float* __restrict__ yv) {
  int row = blockIdx.x, d = threadIdx.x;
  int b = row / CTX;
  int idx = topIdx[row];
  float v = kf[(size_t)b * D_MAIN + d] - bf2f(kib[(size_t)idx * D_MAIN + d]);
  diff[(size_t)row * D_MAIN + d] = f2bf(v);
  if (d == 0) yv[row] = cand_y[idx];
}

__global__ void reduce_v_kernel(const float* __restrict__ Vc, float* __restrict__ Vacc) {
  int b = blockIdx.x, d = threadIdx.x;
  const float* p = Vc + (size_t)b * CTX * D_MAIN + d;
  float s = 0.0f;
  for (int c = 0; c < CTX; ++c) s += p[(size_t)c * D_MAIN];
  Vacc[(size_t)b * D_MAIN + d] = s;
}

__global__ void addv_ln_kernel(const float* __restrict__ x, const float* __restrict__ V,
                               const float* __restrict__ g, const float* __restrict__ beta,
                               float* __restrict__ xr, unsigned short* __restrict__ xnb) {
  __shared__ float red[256];
  int b = blockIdx.x, d = threadIdx.x;
  float val = x[(size_t)b * D_MAIN + d];
  if (V) val += V[(size_t)b * D_MAIN + d];
  if (xr) xr[(size_t)b * D_MAIN + d] = val;
  red[d] = val; __syncthreads();
  for (int s = 128; s > 0; s >>= 1) { if (d < s) red[d] += red[d + s]; __syncthreads(); }
  float mean = red[0] * (1.0f / 256.0f);
  __syncthreads();
  float c = val - mean;
  red[d] = c * c; __syncthreads();
  for (int s = 128; s > 0; s >>= 1) { if (d < s) red[d] += red[d + s]; __syncthreads(); }
  float var = red[0] * (1.0f / 256.0f);
  float nrm = c * rsqrtf(var + 1e-5f);
  xnb[(size_t)b * D_MAIN + d] = f2bf(nrm * g[d] + beta[d]);
}

__global__ void head_kernel(const float* __restrict__ x2, const float* __restrict__ g,
                            const float* __restrict__ beta, const float* __restrict__ Pw,
                            const float* __restrict__ Pb, float* __restrict__ out) {
  __shared__ float red[256];
  int b = blockIdx.x, d = threadIdx.x;
  float val = x2[(size_t)b * D_MAIN + d];
  red[d] = val; __syncthreads();
  for (int s = 128; s > 0; s >>= 1) { if (d < s) red[d] += red[d + s]; __syncthreads(); }
  float mean = red[0] * (1.0f / 256.0f);
  __syncthreads();
  float c = val - mean;
  red[d] = c * c; __syncthreads();
  for (int s = 128; s > 0; s >>= 1) { if (d < s) red[d] += red[d + s]; __syncthreads(); }
  float var = red[0] * (1.0f / 256.0f);
  float h = c * rsqrtf(var + 1e-5f) * g[d] + beta[d];
  h = h > 0.0f ? h : 0.0f;
  __syncthreads();
  red[d] = h * Pw[d]; __syncthreads();
  for (int s = 128; s > 0; s >>= 1) { if (d < s) red[d] += red[d + s]; __syncthreads(); }
  if (d == 0) out[b] = red[0] + Pb[0];
}

// ---------------------------------------------------------------- host
static inline int cdiv(long a, long b) { return (int)((a + b - 1) / b); }

extern "C" void kernel_launch(void* const* d_in, const int* in_sizes, int n_in,
                              void* d_out, int out_size, void* d_ws, size_t ws_size,
                              hipStream_t stream) {
  const float* x_num  = (const float*)d_in[0];
  const float* x_bin  = (const float*)d_in[1];
  const float* c_num  = (const float*)d_in[2];
  const float* c_bin  = (const float*)d_in[3];
  const float* cand_y = (const float*)d_in[4];
  const float* lin_w  = (const float*)d_in[5];
  const float* lin_b  = (const float*)d_in[6];
  const float* K_w    = (const float*)d_in[7];
  const float* K_b    = (const float*)d_in[8];
  const float* Y_w    = (const float*)d_in[9];
  const float* Y_b    = (const float*)d_in[10];
  const float* T_w1   = (const float*)d_in[11];
  const float* T_b1   = (const float*)d_in[12];
  const float* T_w2   = (const float*)d_in[13];
  const float* bp_g   = (const float*)d_in[14];
  const float* bp_be  = (const float*)d_in[15];
  const float* bp_w1  = (const float*)d_in[16];
  const float* bp_b1  = (const float*)d_in[17];
  const float* bp_w2  = (const float*)d_in[18];
  const float* bp_b2  = (const float*)d_in[19];
  const float* P_g    = (const float*)d_in[20];
  const float* P_be   = (const float*)d_in[21];
  const float* P_w    = (const float*)d_in[22];
  const float* P_b    = (const float*)d_in[23];
  (void)n_in; (void)out_size; (void)ws_size;

  const int B  = in_sizes[0] / 96;          // 1024
  const int N  = in_sizes[2] / 96;          // 100000
  const int Npad = ((N + 255) / 256) * 256; // 100096
  const int BC = B * CTX;                   // 98304

  char* w = (char*)d_ws; size_t off = 0;
  auto alloc = [&](size_t bytes) -> void* {
    void* p = w + off; off = (off + bytes + 255) & ~(size_t)255; return p;
  };
  unsigned short* candF   = (unsigned short*)alloc((size_t)N * 128 * 2);
  unsigned short* xF      = (unsigned short*)alloc((size_t)B * 128 * 2);
  unsigned short* lin_wt  = (unsigned short*)alloc((size_t)256 * 128 * 2);
  unsigned short* K_wt    = (unsigned short*)alloc((size_t)256 * 256 * 2);
  unsigned short* T_w1t   = (unsigned short*)alloc((size_t)512 * 256 * 2);
  unsigned short* T_w2t   = (unsigned short*)alloc((size_t)256 * 512 * 2);
  unsigned short* bp_w1t  = (unsigned short*)alloc((size_t)512 * 256 * 2);
  unsigned short* bp_w2t  = (unsigned short*)alloc((size_t)256 * 512 * 2);
  unsigned short* h_cand  = (unsigned short*)alloc((size_t)N * 256 * 2);
  unsigned short* ki_b    = (unsigned short*)alloc((size_t)Npad * 256 * 2);
  float*          ki2     = (float*)alloc((size_t)Npad * 4);
  float*          x_f     = (float*)alloc((size_t)B * 256 * 4);
  unsigned short* x_b     = (unsigned short*)alloc((size_t)B * 256 * 2);
  float*          k_f     = (float*)alloc((size_t)B * 256 * 4);
  unsigned short* k_b     = (unsigned short*)alloc((size_t)B * 256 * 2);
  int*            topIdx  = (int*)alloc((size_t)B * CTX * 4);
  float*          wts     = (float*)alloc((size_t)B * CTX * 4);
  unsigned short* diff_b  = (unsigned short*)alloc((size_t)BC * 256 * 2);
  float*          yv      = (float*)alloc((size_t)BC * 4);
  unsigned short* Hc_b    = (unsigned short*)alloc((size_t)BC * 512 * 2);
  float*          Vc      = (float*)alloc((size_t)BC * 256 * 4);
  float*          Vacc    = (float*)alloc((size_t)B * 256 * 4);
  float*          xr      = (float*)alloc((size_t)B * 256 * 4);
  unsigned short* xn_b    = (unsigned short*)alloc((size_t)B * 256 * 2);
  unsigned short* h1_b    = (unsigned short*)alloc((size_t)B * 512 * 2);
  float*          x2_f    = (float*)alloc((size_t)B * 256 * 4);

  auto launch_gemm = [&](const unsigned short* Am, const unsigned short* Bt, const float* bias,
                         const float* radd, float* outF, unsigned short* outB,
                         int M, int Nn, int Kp, int relu, int mode,
                         const float* wr, const float* yvp, const float* Ywp, const float* Ybp) {
    int MT = (Nn == 256) ? 2 : 1;
    int blocks = (M / 16) / MT;
    size_t sh = (size_t)MT * 16 * (Kp + 8) * 2;
    gemm_bf16_kernel<<<blocks, 256, sh, stream>>>(Am, Bt, bias, radd, outF, outB,
                                                  M, Nn, Kp, relu, mode, wr, yvp, Ywp, Ybp);
  };

  // 1) features -> bf16 (K padded 112 -> 128); weights -> transposed bf16
  pack_feat_kernel<<<cdiv((long)N * 128, 256), 256, 0, stream>>>(c_num, c_bin, candF, N);
  pack_feat_kernel<<<cdiv((long)B * 128, 256), 256, 0, stream>>>(x_num, x_bin, xF, B);
  transpose_bf16_kernel<<<cdiv(256 * 128, 256), 256, 0, stream>>>(lin_w, lin_wt, 112, 256, 128);
  transpose_bf16_kernel<<<cdiv(256 * 256, 256), 256, 0, stream>>>(K_w, K_wt, 256, 256, 256);
  transpose_bf16_kernel<<<cdiv(512 * 256, 256), 256, 0, stream>>>(T_w1, T_w1t, 256, 512, 256);
  transpose_bf16_kernel<<<cdiv(256 * 512, 256), 256, 0, stream>>>(T_w2, T_w2t, 512, 256, 512);
  transpose_bf16_kernel<<<cdiv(512 * 256, 256), 256, 0, stream>>>(bp_w1, bp_w1t, 256, 512, 256);
  transpose_bf16_kernel<<<cdiv(256 * 512, 256), 256, 0, stream>>>(bp_w2, bp_w2t, 512, 256, 512);

  // 2) candidate embedding + keys + |ki|^2
  launch_gemm(candF, lin_wt, lin_b, nullptr, nullptr, h_cand, N, 256, 128, 0, 0,
              nullptr, nullptr, nullptr, nullptr);
  launch_gemm(h_cand, K_wt, K_b, nullptr, nullptr, ki_b, N, 256, 256, 0, 0,
              nullptr, nullptr, nullptr, nullptr);
  rowsumsq_kernel<<<N, 256, 0, stream>>>(ki_b, ki2);

  // 3) batch embedding + keys
  launch_gemm(xF, lin_wt, lin_b, nullptr, x_f, x_b, B, 256, 128, 0, 0,
              nullptr, nullptr, nullptr, nullptr);
  launch_gemm(x_b, K_wt, K_b, nullptr, k_f, k_b, B, 256, 256, 0, 0,
              nullptr, nullptr, nullptr, nullptr);

  // 4) fused dist GEMM + top-96 + softmax weights (ki_b L2-resident: 51MB < 192MB)
  knn_topk_kernel<<<B / 16, 256, 0, stream>>>(k_b, ki_b, ki2, N, Npad, topIdx, wts);

  // 5) diff rows + gathered labels
  build_diff_kernel<<<BC, 256, 0, stream>>>(k_f, ki_b, topIdx, cand_y, diff_b, yv);

  // 6) T-MLP: H = relu(diff@T_w1+T_b1); Vc = w*(encode_y + H@T_w2); V = sum_c Vc
  launch_gemm(diff_b, T_w1t, T_b1, nullptr, nullptr, Hc_b, BC, 512, 256, 1, 0,
              nullptr, nullptr, nullptr, nullptr);
  launch_gemm(Hc_b, T_w2t, nullptr, nullptr, Vc, nullptr, BC, 256, 512, 0, 1,
              wts, yv, Y_w, Y_b);
  reduce_v_kernel<<<B, 256, 0, stream>>>(Vc, Vacc);

  // 7) predictor block: x += V; x += relu(LN(x)@bp_w1+bp_b1)@bp_w2+bp_b2
  addv_ln_kernel<<<B, 256, 0, stream>>>(x_f, Vacc, bp_g, bp_be, xr, xn_b);
  launch_gemm(xn_b, bp_w1t, bp_b1, nullptr, nullptr, h1_b, B, 512, 256, 1, 0,
              nullptr, nullptr, nullptr, nullptr);
  launch_gemm(h1_b, bp_w2t, bp_b2, xr, x2_f, nullptr, B, 256, 512, 0, 0,
              nullptr, nullptr, nullptr, nullptr);

  // 8) head
  head_kernel<<<B, 256, 0, stream>>>(x2_f, P_g, P_be, P_w, P_b, (float*)d_out);
}
